// Net_78434692760179
// MI455X (gfx1250) — matrix operand order repaired and checked by reading, not compiled, not run on hardware
//
#include <hip/hip_runtime.h>

typedef _Float16 h16;
typedef __attribute__((ext_vector_type(8)))  _Float16 v8h;
typedef __attribute__((ext_vector_type(16))) _Float16 v16h;
typedef __attribute__((ext_vector_type(8)))  float    v8f;

#define B_    2048
#define L_    100
#define M_    (B_ * L_)      // 204800 rows, divisible by 128
#define KD    672            // 660 real + 12 zero pad = 21 * 32
#define KREAL 660
#define ND    432            // 420 real + 12 zero pad = 27 * 16
#define NF    140
#define WORD  250
#define POS   80
#define LDA   40             // LDS row stride in halves (32 + 8 pad)
#define ASZ   (128 * LDA)    // 5120 halves per A buffer
#define BSZ   (144 * LDA)    // 5760 halves per B buffer

// ---------------------------------------------------------------------------
// CDNA5 async global->LDS copy (ASYNCcnt-tracked), CDNA5 ISA.
// lds_off = byte offset into LDS (low 32 bits of a generic shared pointer).
// ---------------------------------------------------------------------------
__device__ __forceinline__ void async_b128(unsigned lds_off, const void* g) {
  asm volatile("global_load_async_to_lds_b128 %0, %1, off"
               :: "v"(lds_off), "v"(g) : "memory");
}
__device__ __forceinline__ void wait_async0() {
  asm volatile("s_wait_asynccnt 0x0" ::: "memory");
}

// ---------------------------------------------------------------------------
// Pass 1: build activation matrix X[M][672] in f16
// row = [word(250) | pos1(80) | pos2(80) | sdp*word(250) | 12 zeros]
// One wave per row.
// ---------------------------------------------------------------------------
__global__ void __launch_bounds__(256)
build_x(const int* __restrict__ tok, const int* __restrict__ p1,
        const int* __restrict__ p2, const float* __restrict__ sdp,
        const float* __restrict__ wemb, const float* __restrict__ p1e,
        const float* __restrict__ p2e, h16* __restrict__ X) {
  int wave = threadIdx.x >> 5, lane = threadIdx.x & 31;
  int m = blockIdx.x * 8 + wave;
  if (m >= M_) return;
  int   t  = tok[m];
  int   a1 = p1[m];
  int   a2 = p2[m];
  float s  = sdp[m];
  h16* dst = X + (size_t)m * KD;
  const float* wr = wemb + (size_t)t * WORD;
  for (int i = lane; i < WORD; i += 32) {
    float w = wr[i];
    dst[i]       = (h16)w;
    dst[410 + i] = (h16)(s * w);
  }
  const float* r1 = p1e + (size_t)a1 * POS;
  const float* r2 = p2e + (size_t)a2 * POS;
  for (int i = lane; i < POS; i += 32) {
    dst[250 + i] = (h16)r1[i];
    dst[330 + i] = (h16)r2[i];
  }
  if (lane < 12) dst[660 + lane] = (h16)0.f;
}

// ---------------------------------------------------------------------------
// Pass 1b: repack conv_w (F=140, C=660, T=3) into Bt[n][k], n = t*140 + f,
// K-major per column so B fragments are contiguous 16-half loads.
// ---------------------------------------------------------------------------
__global__ void build_w(const float* __restrict__ cw, h16* __restrict__ Bt) {
  int idx = blockIdx.x * blockDim.x + threadIdx.x;
  const int total = ND * KD;
  for (; idx < total; idx += gridDim.x * blockDim.x) {
    int n = idx / KD, k = idx - n * KD;
    float v = 0.f;
    if (n < 3 * NF && k < KREAL) {
      int t = n / NF, f = n - t * NF;
      v = cw[((size_t)f * KREAL + k) * 3 + t];
    }
    Bt[idx] = (h16)v;
  }
}

// ---------------------------------------------------------------------------
// Pass 2: GEMM  Z[M][432] = X[M][672] * Bt^T  via v_wmma_f32_16x16x32_f16.
// Block: 256 threads (8 wave32). Block tile M=128 (16 rows per wave),
// N in 3 groups of 144 (9 wmma tiles per wave), K in 21 steps of 32.
// Double-buffered LDS filled by async global->LDS; one barrier per K-step.
// All 9 B fragments preloaded per K-step, then 9 WMMAs issued back-to-back.
// ---------------------------------------------------------------------------
__global__ void __launch_bounds__(256)
gemm(const h16* __restrict__ X, const h16* __restrict__ Bt,
     float* __restrict__ Z) {
  __shared__ h16 smem[2 * ASZ + 2 * BSZ];   // 43.5 KB
  h16* const As0 = smem;
  h16* const Bs0 = smem + 2 * ASZ;

  const int tid   = threadIdx.x;
  const int wave  = tid >> 5;
  const int lane  = tid & 31;
  const int mBase = blockIdx.x * 128;
  const int l16   = lane & 15;
  const int hsel  = lane >> 4;            // 0 or 1
  const int arow  = (wave << 4) + l16;    // A row within block tile
  const int kha   = hsel << 3;            // half-offset 0 / 8, unused (A loads use khb)
  const int khb   = hsel << 4;            // B: half-offset 0 / 16

  // staging geometry: each thread moves 32B of A (row tid/2, 16-half part)
  const int arow_st = tid >> 1;
  const int apart   = (tid & 1) << 4;

  auto issue_tiles = [&](int g, int k, int buf) {
    {  // A tile: 128 rows x 32 halves
      const h16* src = X + (size_t)(mBase + arow_st) * KD + k * 32 + apart;
      unsigned l = (unsigned)(uintptr_t)&As0[buf * ASZ + arow_st * LDA + apart];
      async_b128(l,      src);
      async_b128(l + 16, src + 8);
    }
    // B tile: 144 cols x 32 halves (288 tasks of 16 halves)
    for (int task = tid; task < 288; task += 256) {
      int col  = task >> 1;
      int part = (task & 1) << 4;
      const h16* src = Bt + (size_t)(g * 144 + col) * KD + k * 32 + part;
      unsigned l = (unsigned)(uintptr_t)&Bs0[buf * BSZ + col * LDA + part];
      async_b128(l,      src);
      async_b128(l + 16, src + 8);
    }
  };

  for (int g = 0; g < 3; ++g) {
    v8f acc[9];
#pragma unroll
    for (int j = 0; j < 9; ++j)
      acc[j] = (v8f){0.f, 0.f, 0.f, 0.f, 0.f, 0.f, 0.f, 0.f};

    issue_tiles(g, 0, 0);       // prologue (prev group's trailing barrier
    wait_async0();              //  protects buffer 0 against WAR)
    __syncthreads();

#pragma unroll 1
    for (int k = 0; k < 21; ++k) {
      const int cur = k & 1;
      if (k < 20) issue_tiles(g, k + 1, cur ^ 1);   // prefetch next K-step

      const h16* Ab = &As0[cur * ASZ];
      const h16* Bb = &Bs0[cur * BSZ];

      // A fragment: lanes 0-15 -> K 0..15 ; lanes 16-31 -> K 16..31
      v8h a0 = *(const v8h*)(&Ab[arow * LDA + khb]);
      v8h a1 = *(const v8h*)(&Ab[arow * LDA + khb + 8]);
      v16h av = __builtin_shufflevector(a0, a1, 0, 1, 2, 3, 4, 5, 6, 7,
                                        8, 9, 10, 11, 12, 13, 14, 15);
      // preload all 9 B fragments (one clause of ds_loads, partial waits)
      v16h bfrag[9];
#pragma unroll
      for (int j = 0; j < 9; ++j) {
        int col = (j << 4) + l16;
        v8h c0 = *(const v8h*)(&Bb[col * LDA + khb]);
        v8h c1 = *(const v8h*)(&Bb[col * LDA + khb + 8]);
        bfrag[j] = __builtin_shufflevector(c0, c1, 0, 1, 2, 3, 4, 5, 6, 7,
                                           8, 9, 10, 11, 12, 13, 14, 15);
      }
      // 9 back-to-back WMMAs on the matrix pipe
#pragma unroll
      for (int j = 0; j < 9; ++j)
        acc[j] = __builtin_amdgcn_wmma_f32_16x16x32_f16(
            false, av, false, bfrag[j], (short)0, acc[j], false, false);

      wait_async0();      // my async loads (next buffer) have landed
      __syncthreads();    // everyone's loads landed; safe to flip buffers
    }

    // write back: VGPR r holds M = r + 8*hsel, N = lane&15
#pragma unroll
    for (int j = 0; j < 9; ++j) {
      int col = g * 144 + (j << 4) + l16;
#pragma unroll
      for (int r = 0; r < 8; ++r) {
        int row = mBase + (wave << 4) + (hsel << 3) + r;
        Z[(size_t)row * ND + col] = acc[j][r];
      }
    }
  }
}

// ---------------------------------------------------------------------------
// Pass 3: tap-sum (shift by -1/0/+1 rows) + bias + ReLU + max over L,
// then 140->50->11 MLP. One block per batch element.
// ---------------------------------------------------------------------------
__global__ void __launch_bounds__(160)
head(const float* __restrict__ Z, const float* __restrict__ cb,
     const float* __restrict__ w1, const float* __restrict__ b1,
     const float* __restrict__ w2, const float* __restrict__ b2,
     float* __restrict__ out) {
  __shared__ float pooled[NF];
  __shared__ float hmid[50];
  const int b = blockIdx.x;
  const int f = threadIdx.x;
  if (f < NF) {
    const float* zb = Z + (size_t)b * L_ * ND;
    float bias = cb[f];
    float best = -1e30f;
    for (int l = 0; l < L_; ++l) {
      float y = bias;
      if (l > 0)      y += zb[(size_t)(l - 1) * ND + f];          // tap 0
      y += zb[(size_t)l * ND + NF + f];                           // tap 1
      if (l < L_ - 1) y += zb[(size_t)(l + 1) * ND + 2 * NF + f]; // tap 2
      best = fmaxf(best, y);
    }
    pooled[f] = fmaxf(best, 0.f);   // max of relu == relu of max
  }
  __syncthreads();
  if (f < 50) {
    float acc = b1[f];
    for (int i = 0; i < NF; ++i) acc += pooled[i] * w1[i * 50 + f];
    hmid[f] = fmaxf(acc, 0.f);
  }
  __syncthreads();
  if (f < 11) {
    float acc = b2[f];
    for (int j = 0; j < 50; ++j) acc += hmid[j] * w2[j * 11 + f];
    out[b * 11 + f] = acc;
  }
}

// ---------------------------------------------------------------------------
extern "C" void kernel_launch(void* const* d_in, const int* in_sizes, int n_in,
                              void* d_out, int out_size, void* d_ws,
                              size_t ws_size, hipStream_t stream) {
  (void)in_sizes; (void)n_in; (void)out_size; (void)ws_size;
  const int*   tok  = (const int*)d_in[0];
  const int*   p1   = (const int*)d_in[1];
  const int*   p2   = (const int*)d_in[2];
  const float* sdp  = (const float*)d_in[3];
  const float* wemb = (const float*)d_in[4];
  const float* p1e  = (const float*)d_in[5];
  const float* p2e  = (const float*)d_in[6];
  const float* cw   = (const float*)d_in[7];
  const float* cb   = (const float*)d_in[8];
  const float* w1   = (const float*)d_in[9];
  const float* b1   = (const float*)d_in[10];
  const float* w2   = (const float*)d_in[11];
  const float* b2   = (const float*)d_in[12];
  float* out = (float*)d_out;

  h16*   X  = (h16*)d_ws;                          // 204800*672 f16 = 275.3 MB
  h16*   Bt = X + (size_t)M_ * KD;                 // 432*672 f16
  float* Z  = (float*)(Bt + (size_t)ND * KD);      // 204800*432 f32 = 354 MB

  build_x<<<M_ / 8, 256, 0, stream>>>(tok, p1, p2, sdp, wemb, p1e, p2e, X);
  build_w<<<(ND * KD + 255) / 256, 256, 0, stream>>>(cw, Bt);
  gemm<<<M_ / 128, 256, 0, stream>>>(X, Bt, Z);
  head<<<B_, 160, 0, stream>>>(Z, cb, w1, b1, w2, b2, out);
}
